// CrossMatrix_31550829756606
// MI455X (gfx1250) — matrix operand order repaired and checked by reading, not compiled, not run on hardware
//
#include <hip/hip_runtime.h>

// ---------------------------------------------------------------------------
// CrossMatrix: y[b,o] = sum_{i,j} x1[b,i]*x1[b,j]*W[o,i*257+j] + bias[o]
// == GEMM C[2048,256] = A[2048,66049] * W^T with A generated on the fly.
// Compute-bound (69.3 GFLOP vs ~72MB min HBM traffic) -> f16 WMMA, f32 acc.
// Kernel 1: W f32 -> f16, layout [o][i][j padded 257->288], zeros in pad.
// Kernel 2: per-workgroup 64x64 tile; outer loop over i (257), inner 9 K-chunks
//           of 32; A-frag = (x1[b,i] scale) * (resident x1[b,j] base frags);
//           W slices triple-buffered in LDS, prefetch distance 2 via
//           global_load_async_to_lds_b128 + s_wait_asynccnt <= 9
//           (9 async instrs per slice per wave; in-order completion).
// ---------------------------------------------------------------------------

typedef __attribute__((ext_vector_type(16))) _Float16 v16h;
typedef __attribute__((ext_vector_type(8)))  _Float16 v8h;
typedef __attribute__((ext_vector_type(8)))  float    v8f;
typedef int v4i __attribute__((vector_size(16)));

#define FEAT   256
#define FP1    257           // FEAT+1
#define JP     288           // padded inner-j length: 9 chunks of 32
#define NCHUNK 9
#define NOUT   256
#define NBATCH 2048
#define MT     64            // M tile per workgroup
#define NT     64            // N tile per workgroup
#define LSTR   296           // LDS row stride (f16 elems), padded vs 288 for banks
#define NBUF   3             // LDS W-slice buffers (prefetch distance 2)

#if __has_builtin(__builtin_amdgcn_global_load_async_to_lds_b128)
#define ASYNC_LDS 1
#warning "CDNA5 probe: ASYNC staging path (global_load_async_to_lds_b128) ENABLED"
#else
#define ASYNC_LDS 0
#warning "CDNA5 probe: async builtin NOT available - VGPR copy fallback staging"
#endif

__device__ __forceinline__ void copy16_g2l(const _Float16* g, _Float16* l) {
#if ASYNC_LDS
  __builtin_amdgcn_global_load_async_to_lds_b128(
      (__attribute__((address_space(1))) v4i*)(void*)g,
      (__attribute__((address_space(3))) v4i*)(void*)l, 0, 0);
#else
  *(float4*)l = *(const float4*)g;
#endif
}

template <int N>
__device__ __forceinline__ void wait_async() {
#if ASYNC_LDS
#if __has_builtin(__builtin_amdgcn_s_wait_asynccnt)
  __builtin_amdgcn_s_wait_asynccnt(N);
#else
  asm volatile("s_wait_asynccnt %0" ::"i"(N) : "memory");
#endif
#endif
}

// --------------------------- Kernel 1: convert W ---------------------------
// Wh[(o*257 + i)*288 + j] = (j < 257) ? (f16)W[o*66049 + i*257 + j] : 0
__global__ void convert_W_kernel(const float* __restrict__ W,
                                 _Float16* __restrict__ Wh) {
  long long idx = (long long)blockIdx.x * blockDim.x + threadIdx.x;
  const long long total = (long long)NOUT * FP1 * JP;
  if (idx >= total) return;
  int j = (int)(idx % JP);
  long long row = idx / JP;                 // row = o*257 + i
  float v = 0.0f;
  if (j < FP1) v = W[row * (long long)FP1 + j];  // row*257+j == o*66049+i*257+j
  Wh[idx] = (_Float16)v;
}

// --------------------------- Kernel 2: GEMM --------------------------------
__global__ __launch_bounds__(256)
void cross_gemm_kernel(const float* __restrict__ x,
                       const _Float16* __restrict__ Wh,
                       const float* __restrict__ bias,
                       float* __restrict__ out) {
  __shared__ __align__(16) _Float16 lds_x[MT * LSTR];           // x1 tile
  __shared__ __align__(16) _Float16 lds_b[NBUF][NT * LSTR];     // W slices

  const int tid  = threadIdx.x;
  const int lane = tid & 31;
  const int l16  = lane & 15;
  const int wave = tid >> 5;
  const int mw   = wave & 3;        // 4 m-subtiles of 16 rows
  const int nw   = wave >> 2;       // 2 n-halves of 32 cols
  const int m_base = blockIdx.y * MT;
  const int o_base = blockIdx.x * NT;

  // stage W slice for outer index i into lds_b[buf]; 2304 b128 chunks total
  // = exactly 9 async b128 instructions per wave (per thread).
  auto stage_b = [&](int i, int bufsel) {
#pragma unroll
    for (int k = 0; k < NCHUNK; ++k) {
      int idx = k * 256 + tid;            // 0..2303
      int r = idx / 36;                   // o row within tile (36 chunks/row)
      int c = idx - r * 36;
      const _Float16* g =
          Wh + (size_t)((o_base + r) * FP1 + i) * JP + c * 8;
      copy16_g2l(g, &lds_b[bufsel][r * LSTR + c * 8]);
    }
  };

  // prologue: prefetch slices 0 and 1 (async overlaps with x1 staging below)
  stage_b(0, 0);
  stage_b(1, 1);

  // stage padded x1 tile: x1[r][0]=1, x1[r][1..256]=x[row], x1[r][257..287]=0
  for (int q = tid; q < MT * JP; q += 256) {
    int r = q / JP;
    int j = q - r * JP;
    float v;
    if (j == 0)        v = 1.0f;
    else if (j < FP1)  v = x[(size_t)(m_base + r) * FEAT + (j - 1)];
    else               v = 0.0f;
    lds_x[r * LSTR + j] = (_Float16)v;
  }

  wait_async<NCHUNK>();   // slice 0 landed (async completes in issue order)
  __syncthreads();

  // Build the 9 loop-invariant base A fragments (x1[b, j]) in registers.
  // A k order used here (contiguous, same as B): lanes 0-15 -> K c*32+0..15,
  //                        lanes 16-31 -> K c*32+16..31.
  union Frag { v16h v; v8h h[2]; };
  const int arow  = (mw * 16 + l16) * LSTR;
  const int ahalf = (lane < 16) ? 0 : 16;
  v16h a_base[NCHUNK];
#pragma unroll
  for (int c = 0; c < NCHUNK; ++c) {
    Frag f;
    f.h[0] = *(const v8h*)&lds_x[arow + c * 32 + ahalf];
    f.h[1] = *(const v8h*)&lds_x[arow + c * 32 + ahalf + 8];
    a_base[c] = f.v;
  }

  // B k order used here (contiguous, same as A): lanes 0-15 col l16 K=0..15, lanes 16-31 col l16 K=16..31.
  const int bhalf = (lane < 16) ? 0 : 16;
  const int brow0 = (nw * 32 + l16) * LSTR;        // n-tile 0
  const int brow1 = (nw * 32 + 16 + l16) * LSTR;   // n-tile 1

  v8f acc0 = {};
  v8f acc1 = {};

  for (int i = 0; i < FP1; ++i) {
    const int cur = i % NBUF;
    const bool more = (i + 2 < FP1);
    if (more) stage_b(i + 2, (i + 2) % NBUF);   // prefetch distance 2

    const _Float16 s = lds_x[arow + i];  // x1[b_row, i], row-matched to A lanes
    const _Float16* bb = &lds_b[cur][0];

#pragma unroll
    for (int c = 0; c < NCHUNK; ++c) {
      v16h a_s = a_base[c] * s;           // v_pk_mul_f16 x8 (scalar bcast)
      const int jb = c * 32 + bhalf;
      Frag b0, b1;
      b0.h[0] = *(const v8h*)&bb[brow0 + jb];
      b0.h[1] = *(const v8h*)&bb[brow0 + jb + 8];
      b1.h[0] = *(const v8h*)&bb[brow1 + jb];
      b1.h[1] = *(const v8h*)&bb[brow1 + jb + 8];
      acc0 = __builtin_amdgcn_wmma_f32_16x16x32_f16(
          false, a_s, false, b0.v, (short)0, acc0, false, false);
      acc1 = __builtin_amdgcn_wmma_f32_16x16x32_f16(
          false, a_s, false, b1.v, (short)0, acc1, false, false);
    }

    // slice i+1 must be resident before next iteration; newest slice (i+2,
    // 9 async instrs) may stay in flight. Tail: full drain.
    if (more) wait_async<NCHUNK>(); else wait_async<0>();
    __syncthreads();
  }

  // Epilogue: D f32 layout VGPR v -> row v (lanes 0-15) / v+8 (lanes 16-31)
  const int ocol = o_base + nw * 32 + l16;
  const int brow = m_base + mw * 16 + ((lane < 16) ? 0 : 8);
  const float bias0 = bias[ocol];
  const float bias1 = bias[ocol + 16];
#pragma unroll
  for (int v = 0; v < 8; ++v) {
    out[(size_t)(brow + v) * NOUT + ocol]      = acc0[v] + bias0;
    out[(size_t)(brow + v) * NOUT + ocol + 16] = acc1[v] + bias1;
  }
}

// --------------------------- host launcher ---------------------------------
extern "C" void kernel_launch(void* const* d_in, const int* in_sizes, int n_in,
                              void* d_out, int out_size, void* d_ws, size_t ws_size,
                              hipStream_t stream) {
  const float* x    = (const float*)d_in[0];
  const float* W    = (const float*)d_in[1];
  const float* bias = (const float*)d_in[2];
  float* out        = (float*)d_out;
  _Float16* Wh      = (_Float16*)d_ws;   // needs 256*257*288*2 = 37,896,192 B

  const long long total = (long long)NOUT * FP1 * JP;
  const int cblocks = (int)((total + 255) / 256);
  convert_W_kernel<<<cblocks, 256, 0, stream>>>(W, Wh);

  dim3 grid(NOUT / NT, NBATCH / MT);     // (4, 32) = 128 workgroups
  cross_gemm_kernel<<<grid, 256, 0, stream>>>(x, Wh, bias, out);
  (void)in_sizes; (void)n_in; (void)out_size; (void)ws_size;
}
